// ClassicalSelfAttention_65481071400351
// MI455X (gfx1250) — matrix operand order repaired and checked by reading, not compiled, not run on hardware
//
#include <hip/hip_runtime.h>
#include <hip/hip_bf16.h>
#include <cstdint>
#include <cstddef>

#define S_LEN 8192
#define EMB   1024

#define USE_ASYNC_ASM 1

typedef __bf16 bf16;
typedef bf16  v16bf __attribute__((ext_vector_type(16)));
typedef bf16  v8bf  __attribute__((ext_vector_type(8)));
typedef float v8f   __attribute__((ext_vector_type(8)));

// ---- order-preserving f32 <-> u32 encoding for atomic row max ----
__device__ __forceinline__ uint32_t enc_f32(float f) {
    uint32_t b = __float_as_uint(f);
    return (b & 0x80000000u) ? ~b : (b | 0x80000000u);
}
__device__ __forceinline__ float dec_f32(uint32_t e) {
    uint32_t b = (e & 0x80000000u) ? (e & 0x7FFFFFFFu) : ~e;
    return __uint_as_float(b);
}

// ---- async global -> LDS staging (CDNA5, tracked by ASYNCcnt) ----
__device__ __forceinline__ void stage16(bf16* l, const bf16* g) {
#if USE_ASYNC_ASM
    asm volatile("global_load_async_to_lds_b128 %0, %1, off"
                 :: "v"((uint32_t)(uintptr_t)l), "v"((uint64_t)(uintptr_t)g)
                 : "memory");
#else
    *(v8bf*)l = *(const v8bf*)g;
#endif
}
__device__ __forceinline__ void wait_async() {
#if USE_ASYNC_ASM
    asm volatile("s_wait_asynccnt 0x0" ::: "memory");
#endif
}

// ---------------------------------------------------------------
// K0: init row-max (encoded -inf) and row-sum (0)
// ---------------------------------------------------------------
__global__ void init_stats(uint32_t* __restrict__ rowmax_enc,
                           float* __restrict__ rowsum) {
    int i = blockIdx.x * blockDim.x + threadIdx.x;
    if (i < S_LEN) {
        rowmax_enc[i] = enc_f32(-3.0e38f);
        rowsum[i]     = 0.0f;
    }
}

// ---------------------------------------------------------------
// K1: fused f32->bf16 convert + 32x32 LDS-tiled transpose.
//     src: R x C f32. dstN (optional): R x C bf16. dstT: C x R bf16.
//     block (32,8), grid (C/32, R/32)
// ---------------------------------------------------------------
__global__ __launch_bounds__(256)
void transpose_convert(const float* __restrict__ src,
                       bf16* __restrict__ dstN,
                       bf16* __restrict__ dstT,
                       int R, int C) {
    __shared__ bf16 tile[32][33];
    const int c     = blockIdx.x * 32 + threadIdx.x;
    const int rbase = blockIdx.y * 32;
#pragma unroll
    for (int i = 0; i < 4; ++i) {
        int rr = threadIdx.y + i * 8;
        int r  = rbase + rr;
        bf16 b = (bf16)src[(size_t)r * C + c];
        if (dstN) dstN[(size_t)r * C + c] = b;
        tile[rr][threadIdx.x] = b;
    }
    __syncthreads();
#pragma unroll
    for (int i = 0; i < 4; ++i) {
        int rr = threadIdx.y + i * 8;   // column of src tile = row of dstT
        dstT[(size_t)(blockIdx.x * 32 + rr) * R + rbase + threadIdx.x] =
            tile[threadIdx.x][rr];
    }
}

// ---------------------------------------------------------------
// Unified double-buffered LDS GEMM:  C(M x N) = A(M x K) * Bt(N x K)^T
// block = 256 threads (8 waves), block tile 128x128,
// wave tile 32x64 (2x4 WMMA tiles), K-step 32, K % 64 == 0.
// Steady-state loop is branch-free: 2 K-steps with static ping-pong
// buffers; last step pair is peeled.
// MODE 0: C bf16                              (projections)
// MODE 1: C bf16 * 1/sqrt(D), atomic row-max  (scores)
// MODE 2: C f32  * 1/rowsum[row]              (output)
// ---------------------------------------------------------------
template <int MODE>
__global__ __launch_bounds__(256)
void gemm_tt(const bf16* __restrict__ A,
             const bf16* __restrict__ Bt,
             void* __restrict__ Cout,
             int K, int ldc,
             uint32_t* __restrict__ rowmax_enc,
             const float* __restrict__ rowsum) {
    const int tid  = threadIdx.x;
    const int lane = tid & 31;
    const int wave = tid >> 5;
    const int lo   = lane & 15;
    const int hi   = (lane >> 4) & 1;
    const int m0   = blockIdx.y * 128;
    const int n0   = blockIdx.x * 128;
    const int wm   = (wave & 3) * 32;    // 4 wave rows
    const int wn   = (wave >> 2) * 64;   // 2 wave cols

    __shared__ bf16 Asub[2][128][48];    // 32 K-elems, padded to 48 (96B rows)
    __shared__ bf16 Bsub[2][128][48];

    v8f acc[2][4] = {};

    const int srow = tid >> 1;           // staging row 0..127
    const int scol = (tid & 1) * 16;     // staging K-offset 0 / 16

    const bf16* ga = A  + (size_t)(m0 + srow) * K + scol;
    const bf16* gb = Bt + (size_t)(n0 + srow) * K + scol;

    auto stage = [&](int b, int kk) {
        stage16(&Asub[b][srow][scol],     ga + kk);
        stage16(&Asub[b][srow][scol + 8], ga + kk + 8);
        stage16(&Bsub[b][srow][scol],     gb + kk);
        stage16(&Bsub[b][srow][scol + 8], gb + kk + 8);
    };
    auto compute = [&](int b) {
        v16bf afrag[2], bfrag[4];
#pragma unroll
        for (int mt = 0; mt < 2; ++mt) {
            v8bf a0 = *(const v8bf*)&Asub[b][wm + mt * 16 + lo][hi * 8];
            v8bf a1 = *(const v8bf*)&Asub[b][wm + mt * 16 + lo][16 + hi * 8];
            afrag[mt] = __builtin_shufflevector(a0, a1,
                0, 1, 2, 3, 4, 5, 6, 7, 8, 9, 10, 11, 12, 13, 14, 15);
        }
#pragma unroll
        for (int nt = 0; nt < 4; ++nt) {
            v8bf b0 = *(const v8bf*)&Bsub[b][wn + nt * 16 + lo][hi * 8];
            v8bf b1 = *(const v8bf*)&Bsub[b][wn + nt * 16 + lo][16 + hi * 8];
            bfrag[nt] = __builtin_shufflevector(b0, b1,
                0, 1, 2, 3, 4, 5, 6, 7, 8, 9, 10, 11, 12, 13, 14, 15);
        }
#pragma unroll
        for (int mt = 0; mt < 2; ++mt)
#pragma unroll
            for (int nt = 0; nt < 4; ++nt)
                acc[mt][nt] = __builtin_amdgcn_wmma_f32_16x16x32_bf16(
                    false, afrag[mt], false, bfrag[nt], (short)0,
                    acc[mt][nt], false, false);
    };

    // prologue: stage K-step 0 into buffer 0
    stage(0, 0);
    wait_async();
    __syncthreads();

    // steady state: branch-free, 2 K-steps per iteration, static buffers
    int k0 = 0;
    for (; k0 < K - 64; k0 += 64) {
        stage(1, k0 + 32);
        compute(0);
        wait_async();
        __syncthreads();
        stage(0, k0 + 64);
        compute(1);
        wait_async();
        __syncthreads();
    }
    // peeled last pair (k0 == K - 64)
    stage(1, k0 + 32);
    compute(0);
    wait_async();
    __syncthreads();
    compute(1);

    // ---- epilogue ----
    const float scale = (MODE == 1) ? 0.03125f : 1.0f;  // 1/sqrt(1024)
#pragma unroll
    for (int mt = 0; mt < 2; ++mt) {
#pragma unroll
        for (int r = 0; r < 8; ++r) {
            const int row = m0 + wm + mt * 16 + hi * 8 + r;
            float v[4];
#pragma unroll
            for (int nt = 0; nt < 4; ++nt) v[nt] = acc[mt][nt][r] * scale;

            if (MODE == 2) {
                const float inv = 1.0f / rowsum[row];
                float* C = (float*)Cout;
#pragma unroll
                for (int nt = 0; nt < 4; ++nt)
                    C[(size_t)row * ldc + n0 + wn + nt * 16 + lo] = v[nt] * inv;
            } else {
                bf16* C = (bf16*)Cout;
#pragma unroll
                for (int nt = 0; nt < 4; ++nt)
                    C[(size_t)row * ldc + n0 + wn + nt * 16 + lo] = (bf16)v[nt];
                if (MODE == 1) {
                    float m = fmaxf(fmaxf(v[0], v[1]), fmaxf(v[2], v[3]));
                    m = fmaxf(m, __shfl_xor(m, 1, 32));
                    m = fmaxf(m, __shfl_xor(m, 2, 32));
                    m = fmaxf(m, __shfl_xor(m, 4, 32));
                    m = fmaxf(m, __shfl_xor(m, 8, 32));
                    if (lo == 0) atomicMax(&rowmax_enc[row], enc_f32(m));
                }
            }
        }
    }
}

// ---------------------------------------------------------------
// K4: S <- exp(S - rowmax), atomicAdd row sums
// ---------------------------------------------------------------
__global__ __launch_bounds__(256)
void exp_pass(bf16* __restrict__ Sbuf,
              const uint32_t* __restrict__ rowmax_enc,
              float* __restrict__ rowsum) {
    const int row = blockIdx.y;
    const int c0  = (blockIdx.x * 256 + threadIdx.x) * 8;
    const float m = dec_f32(rowmax_enc[row]);

    bf16* p = Sbuf + (size_t)row * S_LEN + c0;
    v8bf d = *(const v8bf*)p;
    v8bf o;
    float s = 0.0f;
#pragma unroll
    for (int i = 0; i < 8; ++i) {
        float e = __expf((float)d[i] - m);
        s += e;
        o[i] = (bf16)e;
    }
    *(v8bf*)p = o;

    s += __shfl_xor(s, 1, 32);
    s += __shfl_xor(s, 2, 32);
    s += __shfl_xor(s, 4, 32);
    s += __shfl_xor(s, 8, 32);
    s += __shfl_xor(s, 16, 32);
    __shared__ float red[8];
    if ((threadIdx.x & 31) == 0) red[threadIdx.x >> 5] = s;
    __syncthreads();
    if (threadIdx.x == 0) {
        float t = 0.0f;
#pragma unroll
        for (int i = 0; i < 8; ++i) t += red[i];
        atomicAdd(&rowsum[row], t);
    }
}

// ---------------------------------------------------------------
extern "C" void kernel_launch(void* const* d_in, const int* in_sizes, int n_in,
                              void* d_out, int out_size, void* d_ws, size_t ws_size,
                              hipStream_t stream) {
    const float* X  = (const float*)d_in[0];
    const float* Wr = (const float*)d_in[1];
    const float* We = (const float*)d_in[2];
    float* Out = (float*)d_out;

    char* ws = (char*)d_ws;
    size_t off = 0;
    bf16* Xb  = (bf16*)(ws + off); off += (size_t)S_LEN * EMB * sizeof(bf16);   // 16 MB
    bf16* XbT = (bf16*)(ws + off); off += (size_t)S_LEN * EMB * sizeof(bf16);   // 16 MB (EMB x S)
    bf16* WrT = (bf16*)(ws + off); off += (size_t)EMB * EMB * sizeof(bf16);     //  2 MB
    bf16* WeT = (bf16*)(ws + off); off += (size_t)EMB * EMB * sizeof(bf16);     //  2 MB
    bf16* Qb  = (bf16*)(ws + off); off += (size_t)S_LEN * EMB * sizeof(bf16);   // 16 MB
    bf16* Kb  = (bf16*)(ws + off); off += (size_t)S_LEN * EMB * sizeof(bf16);   // 16 MB
    uint32_t* rowmax = (uint32_t*)(ws + off); off += (size_t)S_LEN * 4;         // 32 KB
    float*    rowsum = (float*)(ws + off);    off += (size_t)S_LEN * 4;         // 32 KB
    bf16* Sbuf = (bf16*)(ws + off); off += (size_t)S_LEN * S_LEN * sizeof(bf16);// 128 MB

    // K0: stat init
    init_stats<<<(S_LEN + 255) / 256, 256, 0, stream>>>(rowmax, rowsum);

    // K1: convert + transpose  (X -> Xb & XbT; Wr/We -> transposed only)
    dim3 tb(32, 8);
    transpose_convert<<<dim3(EMB / 32, S_LEN / 32), tb, 0, stream>>>(
        X, Xb, XbT, S_LEN, EMB);
    transpose_convert<<<dim3(EMB / 32, EMB / 32), tb, 0, stream>>>(
        Wr, (bf16*)nullptr, WrT, EMB, EMB);
    transpose_convert<<<dim3(EMB / 32, EMB / 32), tb, 0, stream>>>(
        We, (bf16*)nullptr, WeT, EMB, EMB);

    // K2: projections  Q = X Wr, K = X We   (A = Xb, Bt = W^T)
    dim3 gp(EMB / 128, S_LEN / 128);
    gemm_tt<0><<<gp, 256, 0, stream>>>(Xb, WrT, Qb, EMB, EMB, nullptr, nullptr);
    gemm_tt<0><<<gp, 256, 0, stream>>>(Xb, WeT, Kb, EMB, EMB, nullptr, nullptr);

    // K3: scores  S = scale * Q K^T  (Bt = Kb, already K-major)
    dim3 gs(S_LEN / 128, S_LEN / 128);
    gemm_tt<1><<<gs, 256, 0, stream>>>(Qb, Kb, Sbuf, EMB, S_LEN, rowmax, nullptr);

    // K4: exp + row sums
    exp_pass<<<dim3(S_LEN / 2048, S_LEN), 256, 0, stream>>>(Sbuf, rowmax, rowsum);

    // K5: output  O = P X / rowsum  (Bt = XbT)
    gemm_tt<2><<<dim3(EMB / 128, S_LEN / 128), 256, 0, stream>>>(
        Sbuf, XbT, Out, S_LEN, EMB, nullptr, rowsum);
}
